// Gemma3Attention_3212635538116
// MI455X (gfx1250) — matrix operand order repaired and checked by reading, not compiled, not run on hardware
//
#include <hip/hip_runtime.h>

typedef __bf16 bf16_t;
typedef __bf16 v16bf __attribute__((ext_vector_type(16)));
typedef float  v8f   __attribute__((ext_vector_type(8)));

union FragBF { v16bf v; uint4 q[2]; bf16_t e[16]; };

// ---- problem constants ----
constexpr int   CB    = 2;      // batch
constexpr int   CS    = 2048;   // seq
constexpr int   CHID  = 2560;
constexpr int   CNH   = 8;
constexpr int   CNKV  = 4;
constexpr int   CHD   = 256;
constexpr int   CGRP  = CNH / CNKV;   // 2
constexpr float CSCALE = 0.0625f;     // 256^-0.5
constexpr float CEPS   = 1e-6f;
constexpr int   CWIN   = 1024;

// ---- CDNA5 async global->LDS path (guarded; falls back to reg staging) ----
#if defined(__has_builtin)
#if __has_builtin(__builtin_amdgcn_global_load_async_to_lds_b128)
#define HAVE_ASYNC_LDS 1
#endif
#endif

#ifdef HAVE_ASYNC_LDS
typedef int v4i_gcc __attribute__((vector_size(16)));
typedef __attribute__((address_space(1))) v4i_gcc* as1_v4i_p;
typedef __attribute__((address_space(3))) v4i_gcc* as3_v4i_p;
__device__ inline void async_cp16(const bf16_t* g, bf16_t* l) {
  __builtin_amdgcn_global_load_async_to_lds_b128((as1_v4i_p)g, (as3_v4i_p)l, 0, 0);
}
#define ASYNC_WAIT() asm volatile("s_wait_asynccnt 0x0" ::: "memory")
#endif

__device__ inline v8f wmma_bf16(const FragBF& a, const FragBF& b, v8f c) {
  return __builtin_amdgcn_wmma_f32_16x16x32_bf16(
      false, a.v, false, b.v, (short)0, c, false, false);
}

__device__ inline float red_max16(float v) {
#pragma unroll
  for (int m = 1; m <= 8; m <<= 1) v = fmaxf(v, __shfl_xor(v, m, 32));
  return v;
}
__device__ inline float red_sum16(float v) {
#pragma unroll
  for (int m = 1; m <= 8; m <<= 1) v += __shfl_xor(v, m, 32);
  return v;
}
__device__ inline float red_sum32(float v) {
#pragma unroll
  for (int m = 1; m <= 16; m <<= 1) v += __shfl_xor(v, m, 32);
  return v;
}

// ---------------------------------------------------------------------------
// fp32 -> bf16 cast, 8 elems/thread (2x float4 load -> 1x b128 store)
// all callers have cols, dstride, doff multiples of 8
// ---------------------------------------------------------------------------
__global__ void cast_copy_kernel(const float* __restrict__ src, bf16_t* __restrict__ dst,
                                 int rows, int cols, int dstride, int doff) {
  int c8n = cols >> 3;
  int idx = blockIdx.x * 256 + threadIdx.x;
  if (idx >= rows * c8n) return;
  int r = idx / c8n, c = (idx % c8n) * 8;
  const float* s = src + (size_t)r * cols + c;
  float4 f0 = *(const float4*)s;
  float4 f1 = *(const float4*)(s + 4);
  FragBF u;
  u.e[0] = (bf16_t)f0.x; u.e[1] = (bf16_t)f0.y; u.e[2] = (bf16_t)f0.z; u.e[3] = (bf16_t)f0.w;
  u.e[4] = (bf16_t)f1.x; u.e[5] = (bf16_t)f1.y; u.e[6] = (bf16_t)f1.z; u.e[7] = (bf16_t)f1.w;
  *(uint4*)&dst[(size_t)r * dstride + doff + c] = u.q[0];
}

// ---------------------------------------------------------------------------
// bf16 GEMM: C[M][N] (f32) = A[M][K] * B[K][N]
// 128x128x32 tiles, 8 waves, double-buffered LDS, async A-tile copy
// ---------------------------------------------------------------------------
__global__ __launch_bounds__(256)
void gemm_bf16_kernel(const bf16_t* __restrict__ A, const bf16_t* __restrict__ Bm,
                      float* __restrict__ C, int M, int N, int K) {
  __shared__ __align__(16) bf16_t As[2][128 * 40];   // [m][k] row-major, pad 40
  __shared__ __align__(16) bf16_t Bt[2][128 * 40];   // transposed [n][k]

  const int tid  = threadIdx.x;
  const int wave = tid >> 5, lane = tid & 31;
  const int hh   = lane >> 4, ln = lane & 15;
  const int wm   = wave & 3, wn = wave >> 2;
  const int m0   = blockIdx.y * 128, n0 = blockIdx.x * 128;
  const int ntiles = K >> 5;

  const int ar = tid >> 1, akg = (tid & 1) * 16;   // A loader: row / k-group
  const int bkr = tid >> 3, bng = (tid & 7) * 16;  // B loader: k-row / n-group
  const bf16_t* aptr = A + (size_t)(m0 + ar) * K + akg;
  const bf16_t* bptr = Bm + (size_t)bkr * N + n0 + bng;
  const int aLds = ar * 40 + akg;

  v8f vzero = {0.f, 0.f, 0.f, 0.f, 0.f, 0.f, 0.f, 0.f};
  v8f acc[2][4];
#pragma unroll
  for (int sm = 0; sm < 2; ++sm)
#pragma unroll
    for (int sn = 0; sn < 4; ++sn) acc[sm][sn] = vzero;

  // ---- prologue: tile 0 -> buffer 0 ----
  {
    FragBF b0;
    b0.q[0] = *(const uint4*)bptr;
    b0.q[1] = *(const uint4*)(bptr + 8);
#ifdef HAVE_ASYNC_LDS
    async_cp16(aptr, &As[0][aLds]);
    async_cp16(aptr + 8, &As[0][aLds + 8]);
#else
    FragBF a0;
    a0.q[0] = *(const uint4*)aptr;
    a0.q[1] = *(const uint4*)(aptr + 8);
    *(uint4*)&As[0][aLds]     = a0.q[0];
    *(uint4*)&As[0][aLds + 8] = a0.q[1];
#endif
#pragma unroll
    for (int j = 0; j < 16; ++j) Bt[0][(bng + j) * 40 + bkr] = b0.e[j];
#ifdef HAVE_ASYNC_LDS
    ASYNC_WAIT();
#endif
    __syncthreads();
  }

  FragBF bnext;
#ifndef HAVE_ASYNC_LDS
  FragBF anext;
#endif
  for (int it = 0; it < ntiles; ++it) {
    const int cur = it & 1, nxt = cur ^ 1;
    const bool more = (it + 1) < ntiles;
    if (more) {
      // start next tile's global traffic before computing current tile
      const bf16_t* sb = bptr + (size_t)(it + 1) * 32 * N;
      bnext.q[0] = *(const uint4*)sb;
      bnext.q[1] = *(const uint4*)(sb + 8);
      const bf16_t* sa = aptr + (it + 1) * 32;
#ifdef HAVE_ASYNC_LDS
      async_cp16(sa, &As[nxt][aLds]);
      async_cp16(sa + 8, &As[nxt][aLds + 8]);
#else
      anext.q[0] = *(const uint4*)sa;
      anext.q[1] = *(const uint4*)(sa + 8);
#endif
      if (it + 2 < ntiles) {
        __builtin_prefetch(aptr + (it + 2) * 32, 0, 1);
        __builtin_prefetch(bptr + (size_t)(it + 2) * 32 * N, 0, 1);
      }
    }

    // ---- compute current tile ----
    FragBF af[2], bfr[4];
#pragma unroll
    for (int sm = 0; sm < 2; ++sm) {
      int row = (wm * 32 + sm * 16 + ln) * 40;
      af[sm].q[0] = *(const uint4*)&As[cur][row + hh * 8];
      af[sm].q[1] = *(const uint4*)&As[cur][row + 16 + hh * 8];
    }
#pragma unroll
    for (int sn = 0; sn < 4; ++sn) {
      int col = (wn * 64 + sn * 16 + ln) * 40;
      bfr[sn].q[0] = *(const uint4*)&Bt[cur][col + hh * 8];
      bfr[sn].q[1] = *(const uint4*)&Bt[cur][col + 16 + hh * 8];
    }
#pragma unroll
    for (int sm = 0; sm < 2; ++sm)
#pragma unroll
      for (int sn = 0; sn < 4; ++sn)
        acc[sm][sn] = wmma_bf16(af[sm], bfr[sn], acc[sm][sn]);

    if (more) {
#pragma unroll
      for (int j = 0; j < 16; ++j) Bt[nxt][(bng + j) * 40 + bkr] = bnext.e[j];
#ifndef HAVE_ASYNC_LDS
      *(uint4*)&As[nxt][aLds]     = anext.q[0];
      *(uint4*)&As[nxt][aLds + 8] = anext.q[1];
#else
      ASYNC_WAIT();
#endif
    }
    __syncthreads();
  }

  // ---- epilogue: 32-bit offsets -> SGPR-base + voffset stores ----
#pragma unroll
  for (int sm = 0; sm < 2; ++sm)
#pragma unroll
    for (int sn = 0; sn < 4; ++sn)
#pragma unroll
      for (int r = 0; r < 8; ++r) {
        unsigned off = (unsigned)((m0 + wm * 32 + sm * 16 + r + 8 * hh) * N +
                                  n0 + wn * 64 + sn * 16 + ln);
        C[off] = acc[sm][sn][r];
      }
}

// ---------------------------------------------------------------------------
// RMSNorm + RoPE; one wave per head vector (HD=256 -> 8 f32 per lane)
// rotate-half pairs d <-> d+128 live in the same lane (128 = 32*4)
// ---------------------------------------------------------------------------
__global__ __launch_bounds__(256)
void rope_norm_kernel(const float* __restrict__ QKV, const float* __restrict__ cosp,
                      const float* __restrict__ sinp, const float* __restrict__ qnw,
                      const float* __restrict__ knw,
                      bf16_t* __restrict__ Qo, bf16_t* __restrict__ Ko,
                      bf16_t* __restrict__ Vo) {
  const int bs   = blockIdx.x;            // b*S + s
  const int b    = bs / CS, s = bs % CS;
  const int wave = threadIdx.x >> 5, lane = threadIdx.x & 31;
  const float* row = QKV + (size_t)bs * 4096;
  const float* cs  = cosp + (size_t)bs * CHD;
  const float* sn  = sinp + (size_t)bs * CHD;

  // ---- Q: head = wave (0..7) ----
  {
    const float* x = row + wave * CHD;
    float v[8], nn[8];
    float ss = 0.f;
#pragma unroll
    for (int j = 0; j < 8; ++j) { v[j] = x[lane + 32 * j]; ss += v[j] * v[j]; }
    ss = red_sum32(ss);
    float rn = rsqrtf(ss * (1.0f / CHD) + CEPS);
#pragma unroll
    for (int j = 0; j < 8; ++j) {
      int d = lane + 32 * j;
      nn[j] = v[j] * rn * (1.0f + qnw[d]);
    }
    bf16_t* dst = Qo + ((size_t)(b * CNH + wave) * CS + s) * CHD;
#pragma unroll
    for (int j = 0; j < 8; ++j) {
      int d = lane + 32 * j;
      float rot = (j < 4) ? -nn[j + 4] : nn[j - 4];
      dst[d] = (bf16_t)(nn[j] * cs[d] + rot * sn[d]);
    }
  }
  if (wave < 4) {
    // ---- K: head = wave (0..3) ----
    const float* x = row + 2048 + wave * CHD;
    float v[8], nn[8];
    float ss = 0.f;
#pragma unroll
    for (int j = 0; j < 8; ++j) { v[j] = x[lane + 32 * j]; ss += v[j] * v[j]; }
    ss = red_sum32(ss);
    float rn = rsqrtf(ss * (1.0f / CHD) + CEPS);
#pragma unroll
    for (int j = 0; j < 8; ++j) {
      int d = lane + 32 * j;
      nn[j] = v[j] * rn * (1.0f + knw[d]);
    }
    bf16_t* dst = Ko + ((size_t)(b * CNKV + wave) * CS + s) * CHD;
#pragma unroll
    for (int j = 0; j < 8; ++j) {
      int d = lane + 32 * j;
      float rot = (j < 4) ? -nn[j + 4] : nn[j - 4];
      dst[d] = (bf16_t)(nn[j] * cs[d] + rot * sn[d]);
    }
  } else {
    // ---- V: head = wave-4 (0..3), plain cast ----
    int h = wave - 4;
    const float* x = row + 3072 + h * CHD;
    bf16_t* dst = Vo + ((size_t)(b * CNKV + h) * CS + s) * CHD;
#pragma unroll
    for (int j = 0; j < 8; ++j) dst[lane + 32 * j] = (bf16_t)x[lane + 32 * j];
  }
}

// ---------------------------------------------------------------------------
// Flash attention, sliding window 1024, GQA (2 q-heads per kv-head).
// Block = (q-tile of 64, head, batch); 4 waves, 16 queries per wave.
// K tile (async copy) and transposed V tile time-share one LDS buffer.
// ---------------------------------------------------------------------------
__global__ __launch_bounds__(128)
void attn_kernel(const bf16_t* __restrict__ Qb, const bf16_t* __restrict__ Kb,
                 const bf16_t* __restrict__ Vb, bf16_t* __restrict__ AO) {
  const int qt = blockIdx.x, h = blockIdx.y, b = blockIdx.z;
  const int kvh = h / CGRP;
  const int tid = threadIdx.x;
  const int wave = tid >> 5, lane = tid & 31;
  const int hh = lane >> 4, ln = lane & 15;
  const int q0w = qt * 64 + wave * 16;

  // KV buffer: K layout [64][264] (16896 elems) or Vt layout [256][80] (20480)
  __shared__ __align__(16) bf16_t KV[256 * 80];
  __shared__ __align__(16) bf16_t Pl[4 * 16 * 80];   // per-wave P tile [16][80]

  // ---- persistent Q fragments (8 k-chunks of 32 over HD=256) ----
  FragBF qfrag[8];
  {
    const bf16_t* qrow =
        Qb + ((size_t)(b * CNH + h) * CS + q0w + ln) * CHD;
#pragma unroll
    for (int kc = 0; kc < 8; ++kc) {
      qfrag[kc].q[0] = *(const uint4*)(qrow + kc * 32 + hh * 8);
      qfrag[kc].q[1] = *(const uint4*)(qrow + kc * 32 + 16 + hh * 8);
    }
  }

  v8f vzero = {0.f, 0.f, 0.f, 0.f, 0.f, 0.f, 0.f, 0.f};
  v8f oacc[16];
#pragma unroll
  for (int t = 0; t < 16; ++t) oacc[t] = vzero;
  float mprev[8], lsum[8];
#pragma unroll
  for (int r = 0; r < 8; ++r) { mprev[r] = -3.0e38f; lsum[r] = 0.f; }

  int lo = qt * 64 - CWIN + 1;
  if (lo < 0) lo = 0;
  const int kb_lo = lo >> 6;

  const bf16_t* kbase = Kb + (size_t)(b * CNKV + kvh) * CS * CHD;
  const bf16_t* vbase = Vb + (size_t)(b * CNKV + kvh) * CS * CHD;

  for (int kb = kb_lo; kb <= qt; ++kb) {
    __syncthreads();
    // ---- load K tile [64 keys][256 hd] into KV as [key][264] ----
    for (int idx = tid; idx < 64 * 16; idx += 128) {
      int key = idx >> 4, dg = (idx & 15) * 16;
      const bf16_t* src = kbase + (size_t)(kb * 64 + key) * CHD + dg;
      bf16_t* l = &KV[key * 264 + dg];
#ifdef HAVE_ASYNC_LDS
      async_cp16(src, l);
      async_cp16(src + 8, l + 8);
#else
      uint4 x0 = *(const uint4*)src;
      uint4 x1 = *(const uint4*)(src + 8);
      *(uint4*)l       = x0;
      *(uint4*)(l + 8) = x1;
#endif
    }
#ifdef HAVE_ASYNC_LDS
    ASYNC_WAIT();
#endif
    __syncthreads();

    // ---- scores: Q(16x256) * K^T(256x64) ----
    v8f sacc[4] = {vzero, vzero, vzero, vzero};
#pragma unroll
    for (int kc = 0; kc < 8; ++kc) {
#pragma unroll
      for (int nt = 0; nt < 4; ++nt) {
        FragBF bfr;
        const bf16_t* p = &KV[(nt * 16 + ln) * 264 + kc * 32 + hh * 8];
        bfr.q[0] = *(const uint4*)p;
        bfr.q[1] = *(const uint4*)(p + 16);
        sacc[nt] = wmma_bf16(qfrag[kc], bfr, sacc[nt]);
      }
    }

    // ---- mask + online softmax (rows live across 16-lane halves) ----
#pragma unroll
    for (int r = 0; r < 8; ++r) {
      int qg = q0w + r + 8 * hh;
      float rowm = -3.0e38f;
#pragma unroll
      for (int nt = 0; nt < 4; ++nt) {
        int kg = kb * 64 + nt * 16 + ln;
        float sv = sacc[nt][r] * CSCALE;
        bool ok = (kg <= qg) && (qg - kg < CWIN);
        sv = ok ? sv : -1.0e30f;
        sacc[nt][r] = sv;
        rowm = fmaxf(rowm, sv);
      }
      rowm = red_max16(rowm);
      float mn = fmaxf(mprev[r], rowm);
      float al = __expf(mprev[r] - mn);
      mprev[r] = mn;
      float ps = 0.f;
#pragma unroll
      for (int nt = 0; nt < 4; ++nt) {
        float pv = __expf(sacc[nt][r] - mn);
        ps += pv;
        Pl[wave * 1280 + (r + 8 * hh) * 80 + nt * 16 + ln] = (bf16_t)pv;
      }
      ps = red_sum16(ps);
      lsum[r] = lsum[r] * al + ps;
#pragma unroll
      for (int t = 0; t < 16; ++t) oacc[t][r] *= al;
    }

    __syncthreads();
    // ---- load V tile transposed into KV as Vt[d][80] ----
    for (int idx = tid; idx < 64 * 16; idx += 128) {
      int key = idx >> 4, dg = (idx & 15) * 16;
      const bf16_t* src = vbase + (size_t)(kb * 64 + key) * CHD + dg;
      FragBF u;
      u.q[0] = *(const uint4*)src;
      u.q[1] = *(const uint4*)(src + 8);
#pragma unroll
      for (int j = 0; j < 16; ++j) KV[(dg + j) * 80 + key] = u.e[j];
    }
    __syncthreads();

    // ---- O += P(16x64) * V(64x256) ----
#pragma unroll
    for (int kc = 0; kc < 2; ++kc) {
      FragBF pf;
      const bf16_t* pp = &Pl[wave * 1280 + ln * 80 + kc * 32];
      pf.q[0] = *(const uint4*)(pp + hh * 8);
      pf.q[1] = *(const uint4*)(pp + 16 + hh * 8);
#pragma unroll
      for (int t = 0; t < 16; ++t) {
        FragBF vf;
        const bf16_t* vp = &KV[(t * 16 + ln) * 80 + kc * 32 + hh * 8];
        vf.q[0] = *(const uint4*)vp;
        vf.q[1] = *(const uint4*)(vp + 16);
        oacc[t] = wmma_bf16(pf, vf, oacc[t]);
      }
    }
  }

  // ---- normalize and store to [B*S][NH*HD] bf16 (32-bit offsets) ----
#pragma unroll
  for (int r = 0; r < 8; ++r) {
    float inv = 1.0f / lsum[r];
    int qg = q0w + r + 8 * hh;
    unsigned base = (unsigned)((b * CS + qg) * (CNH * CHD) + h * CHD);
#pragma unroll
    for (int t = 0; t < 16; ++t)
      AO[base + t * 16 + ln] = (bf16_t)(oacc[t][r] * inv);
  }
}

// ---------------------------------------------------------------------------
extern "C" void kernel_launch(void* const* d_in, const int* in_sizes, int n_in,
                              void* d_out, int out_size, void* d_ws, size_t ws_size,
                              hipStream_t stream) {
  (void)in_sizes; (void)n_in; (void)out_size; (void)ws_size;
  const float* hs   = (const float*)d_in[0];
  const float* cosp = (const float*)d_in[1];
  const float* sinp = (const float*)d_in[2];
  const float* wq   = (const float*)d_in[3];
  const float* wk   = (const float*)d_in[4];
  const float* wv   = (const float*)d_in[5];
  const float* wo   = (const float*)d_in[6];
  const float* qnw  = (const float*)d_in[7];
  const float* knw  = (const float*)d_in[8];
  float* out = (float*)d_out;
  char* ws = (char*)d_ws;

  const int M = CB * CS;                 // 4096
  // workspace layout (bytes)
  const size_t OFF_X    = 0;             // Xbf     4096x2560 bf16 (20.97 MB)
  const size_t OFF_WQKV = 20971520;      // Wqkv    2560x4096 bf16 (20.97 MB)
  const size_t OFF_WO   = 41943040;      // Wo      2048x2560 bf16 (10.49 MB)
  const size_t OFF_QKV  = 52428800;      // QKV f32 4096x4096      (67.11 MB)
  const size_t OFF_Q    = 119537664;     // Q bf16  2*8*2048*256   (16.78 MB)
  const size_t OFF_K    = 136314880;     // K bf16  2*4*2048*256   ( 8.39 MB)
  const size_t OFF_V    = 144703488;     // V bf16                 ( 8.39 MB)
  const size_t OFF_AO   = OFF_X;         // attn_out reuses X region (dead)

  bf16_t* Xb    = (bf16_t*)(ws + OFF_X);
  bf16_t* Wqkvb = (bf16_t*)(ws + OFF_WQKV);
  bf16_t* Wob   = (bf16_t*)(ws + OFF_WO);
  float*  QKVf  = (float*)(ws + OFF_QKV);
  bf16_t* Qb    = (bf16_t*)(ws + OFF_Q);
  bf16_t* Kb    = (bf16_t*)(ws + OFF_K);
  bf16_t* Vb    = (bf16_t*)(ws + OFF_V);
  bf16_t* AOb   = (bf16_t*)(ws + OFF_AO);

  auto blocks8 = [](long total) { return (int)((total / 8 + 255) / 256); };

  // 1) casts
  cast_copy_kernel<<<blocks8((long)M * CHID), 256, 0, stream>>>(hs, Xb, M, CHID, CHID, 0);
  cast_copy_kernel<<<blocks8((long)CHID * 2048), 256, 0, stream>>>(wq, Wqkvb, CHID, 2048, 4096, 0);
  cast_copy_kernel<<<blocks8((long)CHID * 1024), 256, 0, stream>>>(wk, Wqkvb, CHID, 1024, 4096, 2048);
  cast_copy_kernel<<<blocks8((long)CHID * 1024), 256, 0, stream>>>(wv, Wqkvb, CHID, 1024, 4096, 3072);
  cast_copy_kernel<<<blocks8((long)2048 * CHID), 256, 0, stream>>>(wo, Wob, 2048, CHID, CHID, 0);

  // 2) QKV projection: [4096,2560] x [2560,4096] -> f32
  gemm_bf16_kernel<<<dim3(4096 / 128, M / 128), 256, 0, stream>>>(Xb, Wqkvb, QKVf, M, 4096, CHID);

  // 3) RMSNorm + RoPE -> Q/K/V bf16 [B,H,S,HD]
  rope_norm_kernel<<<M, 256, 0, stream>>>(QKVf, cosp, sinp, qnw, knw, Qb, Kb, Vb);

  // 4) sliding-window flash attention -> AO bf16 [B*S, NH*HD]
  attn_kernel<<<dim3(CS / 64, CNH, CB), 128, 0, stream>>>(Qb, Kb, Vb, AOb);

  // 5) output projection: [4096,2048] x [2048,2560] -> f32 d_out
  gemm_bf16_kernel<<<dim3(CHID / 128, M / 128), 256, 0, stream>>>(AOb, Wob, out, M, CHID, 2048);
}
